// GCN_CLA_43731357008092
// MI455X (gfx1250) — matrix operand order repaired and checked by reading, not compiled, not run on hardware
//
#include <hip/hip_runtime.h>
#include <hip/hip_bf16.h>

typedef __attribute__((ext_vector_type(16))) __bf16 v16bf;
typedef __attribute__((ext_vector_type(8)))  float  v8f;
typedef __attribute__((ext_vector_type(4)))  float  f32x4;

static constexpr int NN      = 16384;
static constexpr int IN_DIM  = 512;
static constexpr int HID_DIM = 32;
static constexpr int OUT_DIM = 16;

// ---------------------------------------------------------------------------
// Pass 1: T1[k][n] = sum_d x[k][d] * W1[d][n] + b1[n]   (bf16 row-major out)
// One wave covers one row k (n = lane). x[k][d] is wave-uniform, W1 coalesced.
// ---------------------------------------------------------------------------
__global__ __launch_bounds__(256) void linear1_kernel(
    const float* __restrict__ x, const float* __restrict__ W1,
    const float* __restrict__ b1, __bf16* __restrict__ T1)
{
    const int t = blockIdx.x * 256 + threadIdx.x;
    const int k = t >> 5;        // row 0..16383
    const int n = t & 31;        // col 0..31
    const float* xr = x + (size_t)k * IN_DIM;
    float acc = b1[n];
#pragma unroll 8
    for (int d = 0; d < IN_DIM; ++d)
        acc = fmaf(xr[d], W1[d * HID_DIM + n], acc);
    T1[(size_t)k * HID_DIM + n] = (__bf16)acc;
}

// ---------------------------------------------------------------------------
// Pass 3: T2[k][n] = sum_j H[k][j] * W2[j][n] + b2[n]   (bf16 row-major out)
// ---------------------------------------------------------------------------
__global__ __launch_bounds__(256) void linear2_kernel(
    const float* __restrict__ H, const float* __restrict__ W2,
    const float* __restrict__ b2, __bf16* __restrict__ T2)
{
    const int t = blockIdx.x * 256 + threadIdx.x;
    const int k = t >> 4;        // row 0..16383
    const int n = t & 15;        // col 0..15
    const float* hr = H + (size_t)k * HID_DIM;
    float acc = b2[n];
#pragma unroll
    for (int j = 0; j < HID_DIM; ++j)
        acc = fmaf(hr[j], W2[j * OUT_DIM + n], acc);
    T2[(size_t)k * OUT_DIM + n] = (__bf16)acc;
}

// ---------------------------------------------------------------------------
// Passes 2 & 4: Out[M, COLS] = (relu?) adj[M, 16384] @ B[16384, COLS]
// bf16 WMMA 16x16x32, f32 accumulate. One wave = one 16-row tile, full K sweep.
//
// A layout (16-bit A 16x32, ISA 7.12.2): lane = M (mod 16); per lane the 16
// bf16 elements are K = khalf+{0..7} then khalf+16+{0..7}, khalf = (lane>=16)*8.
// B layout (K x 16): lane%16 = N; elements 0..15 = K in the same order as A ->
// strided 2-byte loads (stride COLS) from the row-major bf16 B matrix.
// C layout: lane%16 = N, VGPR r -> M = r + 8*(lane>=16).
// ---------------------------------------------------------------------------
template <int COLS, bool RELU>
__global__ __launch_bounds__(256) void adj_gemm_kernel(
    const float* __restrict__ adj,      // [NN, NN] f32
    const __bf16* __restrict__ B,       // [NN, COLS] bf16 row-major
    float* __restrict__ Out)            // [NN, COLS] f32 row-major
{
    constexpr int NT = COLS / 16;       // number of 16-wide N tiles
    const int lane    = threadIdx.x & 31;
    const int wave    = threadIdx.x >> 5;
    const int tile    = blockIdx.x * 8 + wave;   // 0..1023
    const int rowbase = tile * 16;
    const int m       = lane & 15;
    const int khalf   = (lane >> 4) * 8;

    v8f acc[NT] = {};

    const float* arow = adj + (size_t)(rowbase + m) * NN;

    for (int kk = 0; kk < NN; kk += 32) {
        // ---- A tile: 16x32 f32 -> bf16 (streaming, nontemporal: zero reuse)
        const int kb = kk + khalf;
        f32x4 f0 = __builtin_nontemporal_load((const f32x4*)(arow + kb));
        f32x4 f1 = __builtin_nontemporal_load((const f32x4*)(arow + kb + 4));
        f32x4 f2 = __builtin_nontemporal_load((const f32x4*)(arow + kb + 16));
        f32x4 f3 = __builtin_nontemporal_load((const f32x4*)(arow + kb + 20));
        float af[16] = {f0[0], f0[1], f0[2], f0[3],
                        f1[0], f1[1], f1[2], f1[3],
                        f2[0], f2[1], f2[2], f2[3],
                        f3[0], f3[1], f3[2], f3[3]};
        v16bf a;
#pragma unroll
        for (int i = 0; i < 16; ++i) a[i] = (__bf16)af[i];

        // ---- B tiles (hot in L2/WGP$: whole B is <= 1 MB)
        const __bf16* brow = B + (size_t)(kk + khalf) * COLS + m;
#pragma unroll
        for (int t = 0; t < NT; ++t) {
            v16bf b;
#pragma unroll
            for (int i = 0; i < 8; ++i) {
                b[i]     = brow[(size_t)i * COLS + t * 16];
                b[i + 8] = brow[(size_t)(i + 16) * COLS + t * 16];
            }
            acc[t] = __builtin_amdgcn_wmma_f32_16x16x32_bf16(
                false, a, false, b, (short)0, acc[t], false, false);
        }
    }

    // ---- store C (optionally fused relu)
    const int col  = lane & 15;
    const int rofs = (lane >> 4) * 8;
#pragma unroll
    for (int t = 0; t < NT; ++t) {
#pragma unroll
        for (int r = 0; r < 8; ++r) {
            float v = acc[t][r];
            if (RELU) v = fmaxf(v, 0.0f);
            Out[(size_t)(rowbase + r + rofs) * COLS + t * 16 + col] = v;
        }
    }
}

// ---------------------------------------------------------------------------
extern "C" void kernel_launch(void* const* d_in, const int* in_sizes, int n_in,
                              void* d_out, int out_size, void* d_ws, size_t ws_size,
                              hipStream_t stream)
{
    const float* x   = (const float*)d_in[0];   // [16384, 512]
    const float* adj = (const float*)d_in[1];   // [16384, 16384]
    const float* W1  = (const float*)d_in[2];   // [512, 32]
    const float* b1  = (const float*)d_in[3];   // [32]
    const float* W2  = (const float*)d_in[4];   // [32, 16]
    const float* b2  = (const float*)d_in[5];   // [16]
    float* out = (float*)d_out;                 // [16384, 16]

    char* ws = (char*)d_ws;
    __bf16* T1 = (__bf16*)ws;                                       // 1 MB
    float*  H  = (float*)(ws + (size_t)NN * HID_DIM * sizeof(__bf16));   // 2 MB
    __bf16* T2 = (__bf16*)(ws + (size_t)NN * HID_DIM * (sizeof(__bf16) + sizeof(float)));

    // Pass 1: T1 = x @ W1 + b1  (bf16)
    linear1_kernel<<<(NN * HID_DIM) / 256, 256, 0, stream>>>(x, W1, b1, T1);

    // Pass 2: H = relu(adj @ T1)   -- 1 GiB adj stream, bf16 WMMA
    adj_gemm_kernel<HID_DIM, true><<<NN / 16 / 8, 256, 0, stream>>>(adj, T1, H);

    // Pass 3: T2 = H @ W2 + b2  (bf16)
    linear2_kernel<<<(NN * OUT_DIM) / 256, 256, 0, stream>>>(H, W2, b2, T2);

    // Pass 4: out = adj @ T2       -- second 1 GiB adj stream
    adj_gemm_kernel<OUT_DIM, false><<<NN / 16 / 8, 256, 0, stream>>>(adj, T2, out);
}
